// POPSSMambaLayer_81003083203665
// MI455X (gfx1250) — matrix operand order repaired and checked by reading, not compiled, not run on hardware
//
#include <hip/hip_runtime.h>
#include <hip/hip_bf16.h>

typedef __attribute__((ext_vector_type(16))) _Float16 v16h;
typedef __attribute__((ext_vector_type(8)))  _Float16 v8h;
typedef __attribute__((ext_vector_type(8)))  float    v8f;
typedef __attribute__((ext_vector_type(4)))  int      i4;

// Address-space-qualified int4, matching the async builtin's parameter types:
// param1: int4 addrspace(1)* (prints as "int __vector(4) __device__ *"),
// param2: int4 addrspace(3)*
typedef __attribute__((address_space(1))) i4 g_i4;
typedef __attribute__((address_space(3))) i4 l_i4;

#define D_MODEL 1024
#define D_INNER 2048
#define D_STATE 16
#define D_CONV  4
#define DT_RANK 64
#define LSEQ    1024
#define NTOK    2048            // B * L
#define XZ_COLS (2 * D_INNER)   // 4096

// ---------------------------------------------------------------------------
// CDNA5 async global->LDS staging (ASYNCcnt path). Guarded so the file always
// compiles: falls back to synchronous reg->LDS copies if the builtin is absent.
// ---------------------------------------------------------------------------
#if defined(__has_builtin)
#if __has_builtin(__builtin_amdgcn_global_load_async_to_lds_b128)
#define USE_ASYNC_LDS 1
#endif
#endif

__device__ __forceinline__ void stage_b128(const _Float16* g, _Float16* l)
{
#ifdef USE_ASYNC_LDS
    // integer-cast chain sidesteps address-space conversion diagnostics;
    // low 32 bits of a generic LDS address are the LDS offset (ISA 10.2).
    __builtin_amdgcn_global_load_async_to_lds_b128(
        (g_i4*)(unsigned long long)g,
        (l_i4*)(unsigned int)(unsigned long long)l,
        0, 0);
#else
    *(v8h*)l = *(const v8h*)g;
#endif
}

__device__ __forceinline__ void wait_async_lds()
{
#ifdef USE_ASYNC_LDS
#if __has_builtin(__builtin_amdgcn_s_wait_asynccnt)
    __builtin_amdgcn_s_wait_asynccnt(0);
#else
    asm volatile("s_wait_asynccnt 0" ::: "memory");
#endif
#endif
}

// ---------------------------------------------------------------------------
// LayerNorm over D_MODEL, output f16 for the WMMA GEMM.
// One block per token (2048 blocks x 256 threads), each thread owns 4 floats.
// ---------------------------------------------------------------------------
__global__ __launch_bounds__(256)
void layernorm_f16_kernel(const float* __restrict__ h,
                          const float* __restrict__ w,
                          const float* __restrict__ b,
                          _Float16* __restrict__ out)
{
    __shared__ float red[256];
    const int t = blockIdx.x;
    const float4 v = ((const float4*)(h + (size_t)t * D_MODEL))[threadIdx.x];

    red[threadIdx.x] = v.x + v.y + v.z + v.w;
    __syncthreads();
    for (int off = 128; off > 0; off >>= 1) {
        if (threadIdx.x < off) red[threadIdx.x] += red[threadIdx.x + off];
        __syncthreads();
    }
    const float mu = red[0] * (1.0f / D_MODEL);
    __syncthreads();

    const float dx = v.x - mu, dy = v.y - mu, dz = v.z - mu, dw = v.w - mu;
    red[threadIdx.x] = dx*dx + dy*dy + dz*dz + dw*dw;
    __syncthreads();
    for (int off = 128; off > 0; off >>= 1) {
        if (threadIdx.x < off) red[threadIdx.x] += red[threadIdx.x + off];
        __syncthreads();
    }
    const float rs = rsqrtf(red[0] * (1.0f / D_MODEL) + 1e-5f);

    const float4 wv = ((const float4*)w)[threadIdx.x];
    const float4 bv = ((const float4*)b)[threadIdx.x];
    _Float16* op = out + (size_t)t * D_MODEL + threadIdx.x * 4;
    op[0] = (_Float16)(dx * rs * wv.x + bv.x);
    op[1] = (_Float16)(dy * rs * wv.y + bv.y);
    op[2] = (_Float16)(dz * rs * wv.z + bv.z);
    op[3] = (_Float16)(dw * rs * wv.w + bv.w);
}

// ---------------------------------------------------------------------------
// Transpose fp32 weight W[K][N] -> f16 WT[Npad][K]   (zero pad n >= N)
// ---------------------------------------------------------------------------
__global__ void convert_transpose_kernel(const float* __restrict__ W,
                                         _Float16* __restrict__ WT,
                                         int K, int N, int Npad)
{
    int idx = blockIdx.x * blockDim.x + threadIdx.x;
    if (idx >= Npad * K) return;
    int n = idx / K;
    int k = idx - n * K;
    float v = (n < N) ? W[(size_t)k * N + n] : 0.0f;
    WT[idx] = (_Float16)v;
}

// ---------------------------------------------------------------------------
// WMMA GEMM:  C[M x Nreal] (f32) = A[M x K] (f16, row-major, lda=K)
//                                 x WT[Npad x K] (f16, B transposed, ldb=K)
// Block = 256 thr = 8 waves, block tile 128x64, wave tile 32x32 (2x2 WMMA).
// Tiles double-buffered in LDS via async global->LDS (ASYNCcnt) copies.
// Fragments use contiguous k = 16*half + s for both A and B (same order on both).
// ---------------------------------------------------------------------------
__global__ __launch_bounds__(256)
void gemm_wmma_f16_kernel(const _Float16* __restrict__ A,
                          const _Float16* __restrict__ BT,
                          float* __restrict__ C,
                          int K, int Nreal)
{
    __shared__ __attribute__((aligned(32))) _Float16 As[2][128 * 32]; // 8 KB per buf
    __shared__ __attribute__((aligned(32))) _Float16 Bs[2][64 * 32];  // 4 KB per buf

    const int tid  = threadIdx.x;
    const int lane = tid & 31;
    const int wave = tid >> 5;
    const int half = lane >> 4;   // 0: lanes 0-15, 1: lanes 16-31
    const int l16  = lane & 15;

    const int rowBase = blockIdx.y * 128;
    const int colBase = blockIdx.x * 64;

    // Staging assignments: A tile 128x32 = 512 b128-chunks (2/thread),
    //                      B tile  64x32 = 256 b128-chunks (1/thread).
    const int ar0 = tid >> 2;           // 0..63
    const int ar1 = (tid >> 2) + 64;    // 64..127
    const int co  = (tid & 3) * 8;      // 8-half chunk offset within 32-half row

    v8f acc00 = {0,0,0,0,0,0,0,0};
    v8f acc01 = {0,0,0,0,0,0,0,0};
    v8f acc10 = {0,0,0,0,0,0,0,0};
    v8f acc11 = {0,0,0,0,0,0,0,0};

    // LDS row offsets (in halves) of this wave's fragment rows
    const int amRow0 = (((wave >> 1) << 5) + l16) * 32;
    const int amRow1 = (((wave >> 1) << 5) + 16 + l16) * 32;
    const int bnRow0 = (((wave & 1) << 5) + l16) * 32;
    const int bnRow1 = (((wave & 1) << 5) + 16 + l16) * 32;

    const int KT = K >> 5;

    // prologue: stage tile 0 into buffer 0
    {
        const int k0 = 0;
        stage_b128(A  + (size_t)(rowBase + ar0) * K + k0 + co, &As[0][ar0 * 32 + co]);
        stage_b128(A  + (size_t)(rowBase + ar1) * K + k0 + co, &As[0][ar1 * 32 + co]);
        stage_b128(BT + (size_t)(colBase + ar0) * K + k0 + co, &Bs[0][ar0 * 32 + co]);
    }

    for (int kt = 0; kt < KT; kt++) {
        wait_async_lds();      // this wave's staged tile has landed in LDS
        __syncthreads();       // all waves' tiles have landed / prev reads done

        if (kt + 1 < KT) {     // stage next tile into the other buffer
            const int nb = (kt + 1) & 1;
            const int k0 = (kt + 1) << 5;
            stage_b128(A  + (size_t)(rowBase + ar0) * K + k0 + co, &As[nb][ar0 * 32 + co]);
            stage_b128(A  + (size_t)(rowBase + ar1) * K + k0 + co, &As[nb][ar1 * 32 + co]);
            stage_b128(BT + (size_t)(colBase + ar0) * K + k0 + co, &Bs[nb][ar0 * 32 + co]);
        }

        const _Float16* as = As[kt & 1];
        const _Float16* bs = Bs[kt & 1];

        // A fragment: lanes 0-15 hold K {0..15}; lanes 16-31 {16..31}
        v8h a0lo = *(const v8h*)(as + amRow0 + 16 * half);
        v8h a0hi = *(const v8h*)(as + amRow0 + 8 + 16 * half);
        v8h a1lo = *(const v8h*)(as + amRow1 + 16 * half);
        v8h a1hi = *(const v8h*)(as + amRow1 + 8 + 16 * half);
        v16h a0 = __builtin_shufflevector(a0lo, a0hi, 0,1,2,3,4,5,6,7,8,9,10,11,12,13,14,15);
        v16h a1 = __builtin_shufflevector(a1lo, a1hi, 0,1,2,3,4,5,6,7,8,9,10,11,12,13,14,15);
        // B fragment: lane's column, K = 16*half + (0..15), contiguous in LDS row
        v16h b0 = *(const v16h*)(bs + bnRow0 + 16 * half);
        v16h b1 = *(const v16h*)(bs + bnRow1 + 16 * half);

        acc00 = __builtin_amdgcn_wmma_f32_16x16x32_f16(false, a0, false, b0, (short)0, acc00, false, false);
        acc01 = __builtin_amdgcn_wmma_f32_16x16x32_f16(false, a0, false, b1, (short)0, acc01, false, false);
        acc10 = __builtin_amdgcn_wmma_f32_16x16x32_f16(false, a1, false, b0, (short)0, acc10, false, false);
        acc11 = __builtin_amdgcn_wmma_f32_16x16x32_f16(false, a1, false, b1, (short)0, acc11, false, false);

        __syncthreads();       // done reading this buffer before it is restaged
    }

    // C/D layout: lane holds col N=l16 (+tile), VGPR r -> row M = r + 8*half
    const int M0 = rowBase + ((wave >> 1) << 5);
    const int N0 = colBase + ((wave & 1) << 5);
    v8f accs[2][2] = {{acc00, acc01}, {acc10, acc11}};
    #pragma unroll
    for (int mi = 0; mi < 2; mi++) {
        #pragma unroll
        for (int ni = 0; ni < 2; ni++) {
            const int col = N0 + 16 * ni + l16;
            if (col < Nreal) {
                #pragma unroll
                for (int r = 0; r < 8; r++) {
                    const int row = M0 + 16 * mi + 8 * half + r;
                    C[(size_t)row * Nreal + col] = accs[mi][ni][r];
                }
            }
        }
    }
}

// ---------------------------------------------------------------------------
// Causal depthwise conv (k=4) + SiLU.  x = xz[:, 0:D_INNER].
// ---------------------------------------------------------------------------
__global__ void conv_silu_kernel(const float* __restrict__ xz,
                                 const float* __restrict__ cw,
                                 const float* __restrict__ cb,
                                 float* __restrict__ xact,
                                 _Float16* __restrict__ xact16)
{
    int idx = blockIdx.x * blockDim.x + threadIdx.x;
    if (idx >= NTOK * D_INNER) return;
    const int d   = idx & (D_INNER - 1);
    const int tok = idx >> 11;
    const int b   = tok >> 10;
    const int l   = tok & (LSEQ - 1);

    float acc = cb[d];
    #pragma unroll
    for (int t = 0; t < D_CONV; t++) {
        const int li = l + t - (D_CONV - 1);
        if (li >= 0)
            acc += cw[d * D_CONV + t] * xz[(size_t)(b * LSEQ + li) * XZ_COLS + d];
    }
    const float s = acc / (1.0f + expf(-acc));
    xact[idx]   = s;
    xact16[idx] = (_Float16)s;
}

// dt_r (first 64 cols of xp) -> f16 for the dt_proj GEMM
__global__ void dtr_f16_kernel(const float* __restrict__ xp, _Float16* __restrict__ dtr16)
{
    int idx = blockIdx.x * blockDim.x + threadIdx.x;
    if (idx >= NTOK * DT_RANK) return;
    const int tok = idx >> 6;
    const int j   = idx & (DT_RANK - 1);
    dtr16[idx] = (_Float16)xp[(size_t)tok * (DT_RANK + 2 * D_STATE) + j];
}

// delta = softplus(delta_raw + dt_proj_b)   (in place)
__global__ void softplus_bias_kernel(float* __restrict__ delta, const float* __restrict__ bias)
{
    int idx = blockIdx.x * blockDim.x + threadIdx.x;
    if (idx >= NTOK * D_INNER) return;
    const int d = idx & (D_INNER - 1);
    const float v = delta[idx] + bias[d];
    delta[idx] = fmaxf(v, 0.0f) + log1pf(expf(-fabsf(v)));
}

// ---------------------------------------------------------------------------
// Selective scan: one thread per (b, d) channel; 16-state recurrence over L.
// xp row: [0:64)=dt_r, [64:80)=Bm, [80:96)=Cm
// ---------------------------------------------------------------------------
__global__ __launch_bounds__(256)
void scan_kernel(const float* __restrict__ delta,
                 const float* __restrict__ xact,
                 const float* __restrict__ xp,
                 const float* __restrict__ A_log,
                 const float* __restrict__ D_param,
                 float* __restrict__ yscan)
{
    int idx = blockIdx.x * blockDim.x + threadIdx.x;
    if (idx >= 2 * D_INNER) return;
    const int b = idx / D_INNER;
    const int d = idx - b * D_INNER;

    float Af[D_STATE];
    #pragma unroll
    for (int n = 0; n < D_STATE; n++) Af[n] = -expf(A_log[d * D_STATE + n]);
    const float Dp = D_param[d];

    float h[D_STATE];
    #pragma unroll
    for (int n = 0; n < D_STATE; n++) h[n] = 0.0f;

    for (int l = 0; l < LSEQ; l++) {
        const int tok = b * LSEQ + l;
        const float dt = delta[(size_t)tok * D_INNER + d];
        const float u  = xact [(size_t)tok * D_INNER + d];
        const float* xpt = xp + (size_t)tok * (DT_RANK + 2 * D_STATE);
        float y = 0.0f;
        const float dtu = dt * u;
        #pragma unroll
        for (int n = 0; n < D_STATE; n++) {
            const float dA = expf(dt * Af[n]);
            h[n] = dA * h[n] + dtu * xpt[DT_RANK + n];
            y += h[n] * xpt[DT_RANK + D_STATE + n];
        }
        yscan[(size_t)tok * D_INNER + d] = y + u * Dp;
    }
}

// y_gated16 = f16( yscan * silu(z) ),  z = xz[:, D_INNER:2*D_INNER]
__global__ void gate_kernel(const float* __restrict__ yscan,
                            const float* __restrict__ xz,
                            _Float16* __restrict__ yg16)
{
    int idx = blockIdx.x * blockDim.x + threadIdx.x;
    if (idx >= NTOK * D_INNER) return;
    const int d   = idx & (D_INNER - 1);
    const int tok = idx >> 11;
    const float z = xz[(size_t)tok * XZ_COLS + D_INNER + d];
    const float g = z / (1.0f + expf(-z));
    yg16[idx] = (_Float16)(yscan[idx] * g);
}

// ---------------------------------------------------------------------------
// Launch
// ---------------------------------------------------------------------------
extern "C" void kernel_launch(void* const* d_in, const int* in_sizes, int n_in,
                              void* d_out, int out_size, void* d_ws, size_t ws_size,
                              hipStream_t stream)
{
    const float* hidden    = (const float*)d_in[0];
    const float* norm_w    = (const float*)d_in[1];
    const float* norm_b    = (const float*)d_in[2];
    const float* in_proj_w = (const float*)d_in[3];   // [1024, 4096]
    const float* conv_w    = (const float*)d_in[4];   // [2048, 4]
    const float* conv_b    = (const float*)d_in[5];
    const float* x_proj_w  = (const float*)d_in[6];   // [2048, 96]
    const float* dt_proj_w = (const float*)d_in[7];   // [64, 2048]
    const float* dt_proj_b = (const float*)d_in[8];
    const float* A_log     = (const float*)d_in[9];   // [2048, 16]
    const float* D_param   = (const float*)d_in[10];
    const float* out_proj_w= (const float*)d_in[11];  // [2048, 1024]
    float* out = (float*)d_out;

    // Workspace layout (all offsets multiples of 256B)
    char* ws = (char*)d_ws;
    size_t off = 0;
    _Float16* hn16   = (_Float16*)(ws + off); off += (size_t)NTOK * D_MODEL * 2;        // 4 MB
    _Float16* wt_in  = (_Float16*)(ws + off); off += (size_t)XZ_COLS * D_MODEL * 2;     // 8 MB
    float*    xz     = (float*)   (ws + off); off += (size_t)NTOK * XZ_COLS * 4;        // 32 MB
    float*    xact   = (float*)   (ws + off); off += (size_t)NTOK * D_INNER * 4;        // 16 MB
    _Float16* xact16 = (_Float16*)(ws + off); off += (size_t)NTOK * D_INNER * 2;        // 8 MB
    _Float16* wt_xp  = (_Float16*)(ws + off); off += (size_t)128 * D_INNER * 2;         // 0.5 MB (N=96 padded to 128)
    float*    xp     = (float*)   (ws + off); off += (size_t)NTOK * 96 * 4;             // 0.75 MB
    _Float16* dtr16  = (_Float16*)(ws + off); off += (size_t)NTOK * DT_RANK * 2;        // 0.25 MB
    _Float16* wt_dt  = (_Float16*)(ws + off); off += (size_t)D_INNER * DT_RANK * 2;     // 0.25 MB
    float*    delta  = (float*)   (ws + off); off += (size_t)NTOK * D_INNER * 4;        // 16 MB
    float*    yscan  = (float*)   (ws + off); off += (size_t)NTOK * D_INNER * 4;        // 16 MB
    _Float16* yg16   = (_Float16*)(ws + off); off += (size_t)NTOK * D_INNER * 2;        // 8 MB
    _Float16* wt_out = (_Float16*)(ws + off); off += (size_t)D_MODEL * D_INNER * 2;     // 4 MB

    const int TB = 256;
    const int nElem = NTOK * D_INNER;            // 4.19M
    const int nBlkElem = (nElem + TB - 1) / TB;

    // 1. LayerNorm -> f16
    layernorm_f16_kernel<<<NTOK, TB, 0, stream>>>(hidden, norm_w, norm_b, hn16);

    // 2. Weight transposes/conversions (independent of 1)
    convert_transpose_kernel<<<(XZ_COLS * D_MODEL + TB - 1) / TB, TB, 0, stream>>>(in_proj_w,  wt_in,  D_MODEL, XZ_COLS, XZ_COLS);
    convert_transpose_kernel<<<(128 * D_INNER     + TB - 1) / TB, TB, 0, stream>>>(x_proj_w,   wt_xp,  D_INNER, 96,      128);
    convert_transpose_kernel<<<(D_INNER * DT_RANK + TB - 1) / TB, TB, 0, stream>>>(dt_proj_w,  wt_dt,  DT_RANK, D_INNER, D_INNER);
    convert_transpose_kernel<<<(D_MODEL * D_INNER + TB - 1) / TB, TB, 0, stream>>>(out_proj_w, wt_out, D_INNER, D_MODEL, D_MODEL);

    // 3. xz = hn @ in_proj_w        [2048 x 1024] x [1024 x 4096]
    gemm_wmma_f16_kernel<<<dim3(XZ_COLS / 64, NTOK / 128), TB, 0, stream>>>(hn16, wt_in, xz, D_MODEL, XZ_COLS);

    // 4. depthwise conv + SiLU
    conv_silu_kernel<<<nBlkElem, TB, 0, stream>>>(xz, conv_w, conv_b, xact, xact16);

    // 5. xp = x @ x_proj_w          [2048 x 2048] x [2048 x 96]  (Npad=128)
    gemm_wmma_f16_kernel<<<dim3(128 / 64, NTOK / 128), TB, 0, stream>>>(xact16, wt_xp, xp, D_INNER, 96);

    // 6. dt_r -> f16 ; delta_raw = dt_r @ dt_proj_w ; softplus(+bias)
    dtr_f16_kernel<<<(NTOK * DT_RANK + TB - 1) / TB, TB, 0, stream>>>(xp, dtr16);
    gemm_wmma_f16_kernel<<<dim3(D_INNER / 64, NTOK / 128), TB, 0, stream>>>(dtr16, wt_dt, delta, DT_RANK, D_INNER);
    softplus_bias_kernel<<<nBlkElem, TB, 0, stream>>>(delta, dt_proj_b);

    // 7. selective scan (one thread per (b,d) channel)
    scan_kernel<<<(2 * D_INNER + TB - 1) / TB, TB, 0, stream>>>(delta, xact, xp, A_log, D_param, yscan);

    // 8. gate with silu(z), to f16
    gate_kernel<<<nBlkElem, TB, 0, stream>>>(yscan, xz, yg16);

    // 9. out = y_gated @ out_proj_w  [2048 x 2048] x [2048 x 1024] -> d_out (f32)
    gemm_wmma_f16_kernel<<<dim3(D_MODEL / 64, NTOK / 128), TB, 0, stream>>>(yg16, wt_out, out, D_INNER, D_MODEL);
}
